// MultiHeadAttention_41351945126178
// MI455X (gfx1250) — matrix operand order repaired and checked by reading, not compiled, not run on hardware
//
#include <hip/hip_runtime.h>
#include <hip/hip_bf16.h>
#include <math.h>

#define D_MODEL 1024
#define N_HEADS 16
#define D_K     64
#define BATCH   2
#define SEQ     2048
#define M_TOT   (BATCH*SEQ)   // 4096 tokens

typedef __attribute__((ext_vector_type(16))) __bf16 bf16x16;
typedef __attribute__((ext_vector_type(8)))  __bf16 bf16x8;
typedef __attribute__((ext_vector_type(8)))  float  f32x8;

// Build a 16-bit WMMA A-fragment from two contiguous 8-element chunks.
// CDNA5 16-bit A 16x32 layout: lanes 0-15 hold row M=lane with K = {0..7,16..23},
// lanes 16-31 hold row M=lane-16 with K = {8..15,24..31}. Caller passes the two
// per-lane chunk base pointers (already offset by hi*16, second at +8), so K = 16*hi + s.
static __device__ __forceinline__ bf16x16 make_a_frag(const __bf16* p0, const __bf16* p1) {
  bf16x8 c0 = *(const bf16x8*)p0;
  bf16x8 c1 = *(const bf16x8*)p1;
  bf16x16 r;
#pragma unroll
  for (int i = 0; i < 8; ++i) { r[i] = c0[i]; r[i + 8] = c1[i]; }
  return r;
}

static __device__ __forceinline__ f32x8 wmma_bf16(bf16x16 a, bf16x16 b, f32x8 c) {
  return __builtin_amdgcn_wmma_f32_16x16x32_bf16(false, a, false, b, (short)0, c, false, false);
}

// ---------------------------------------------------------------- fp32 -> bf16
__global__ __launch_bounds__(256) void cvt_bf16_kernel(const float* __restrict__ in,
                                                       __bf16* __restrict__ out, int n) {
  int i = blockIdx.x * blockDim.x + threadIdx.x;
  if (i < n) out[i] = (__bf16)in[i];
}

// --------------------------------------- transposed additive mask bias (fp32)
// maskT[k*S + q] = (mask[q*S + k] == 0) ? -1e9 : 0
// This layout coincides with the WMMA C-fragment layout of the score tile, so
// the attention kernel feeds it directly as the WMMA accumulator init.
__global__ __launch_bounds__(256) void maskbias_kernel(const int* __restrict__ mask,
                                                       float* __restrict__ maskT) {
  int i = blockIdx.x * blockDim.x + threadIdx.x;   // i = k*S + q
  int k = i >> 11, q = i & (SEQ - 1);
  maskT[i] = (mask[(size_t)q * SEQ + k] == 0) ? -1e9f : 0.f;
}

// ------------------------------------------------- WMMA GEMM: C = A * W^T + b
// A: [4096, 1024] bf16 row-major; W: [1024, 1024] bf16 row-major (out-dim major,
// so a row of W is the k-contiguous vector needed for the B fragment column).
// Each wave computes a 32(M) x 64(N) tile: 32 k-steps x 8 WMMAs = 256 WMMAs.
// MODE 0: Q*0.125 -> bf16 [b,h,s,dk]   MODE 1: K -> bf16 [b,h,s,dk]
// MODE 2: V -> bf16 [b,h,dk,s]         MODE 3: out -> f32 [b,s,e]
template <int MODE>
__global__ __launch_bounds__(256) void proj_kernel(const __bf16* __restrict__ A,
                                                   const __bf16* __restrict__ W,
                                                   const float* __restrict__ bias,
                                                   void* __restrict__ outp) {
  const int wid  = threadIdx.x >> 5;
  const int lane = threadIdx.x & 31;
  const int lrow = lane & 15;
  const int hi   = lane >> 4;
  const int t    = blockIdx.x * 8 + wid;   // 2048 wave tiles
  const int m0   = (t >> 4) << 5;          // 32-row token tile
  const int n0   = (t & 15) << 6;          // 64-wide output tile

  f32x8 acc[2][4] = {};
  float bj[4];
#pragma unroll
  for (int j = 0; j < 4; ++j) bj[j] = bias[n0 + j * 16 + lrow];

  const __bf16* arow0 = A + (size_t)(m0 + lrow) * D_MODEL + hi * 16;
  const __bf16* arow1 = arow0 + (size_t)16 * D_MODEL;
  const __bf16* wbase = W + hi * 16;

#pragma unroll 2
  for (int k0 = 0; k0 < D_MODEL; k0 += 32) {
    bf16x16 av0 = make_a_frag(arow0 + k0, arow0 + k0 + 8);
    bf16x16 av1 = make_a_frag(arow1 + k0, arow1 + k0 + 8);
#pragma unroll
    for (int j = 0; j < 4; ++j) {
      const __bf16* wrow = wbase + (size_t)(n0 + j * 16 + lrow) * D_MODEL + k0;
      bf16x16 bv = *(const bf16x16*)wrow;
      acc[0][j] = wmma_bf16(av0, bv, acc[0][j]);
      acc[1][j] = wmma_bf16(av1, bv, acc[1][j]);
    }
  }

  // C fragment: VGPR r -> row +r (lanes 0-15) / +r+8 (lanes 16-31), col = lane%16.
#pragma unroll
  for (int mi = 0; mi < 2; ++mi) {
#pragma unroll
    for (int j = 0; j < 4; ++j) {
      const int n = n0 + j * 16 + lrow;
#pragma unroll
      for (int r = 0; r < 8; ++r) {
        const int m = m0 + mi * 16 + r + hi * 8;
        float v = acc[mi][j][r] + bj[j];
        if (MODE == 3) {
          ((float*)outp)[(size_t)m * D_MODEL + n] = v;
        } else {
          const int b = m >> 11, s = m & (SEQ - 1);
          const int h = n >> 6,  dk = n & (D_K - 1);
          __bf16* o = (__bf16*)outp;
          if (MODE == 2)
            o[((size_t)(b * N_HEADS + h) * D_K + dk) * SEQ + s] = (__bf16)v;              // V^T
          else if (MODE == 0)
            o[((size_t)(b * N_HEADS + h) * SEQ + s) * D_K + dk] = (__bf16)(v * 0.125f);   // Q/sqrt(dk)
          else
            o[((size_t)(b * N_HEADS + h) * SEQ + s) * D_K + dk] = (__bf16)v;              // K
        }
      }
    }
  }
}

// ----------------------------------------------------- flash attention (bf16)
// One wave = 16 queries of one (b,h). Online softmax over 64-key tiles:
// 16 WMMAs per tile (8 scores + 8 P*V). The mask bias is injected for free as
// the C operand of the first score WMMA (maskT layout == C-fragment layout).
__global__ __launch_bounds__(256) void attn_kernel(const __bf16* __restrict__ Qb,
                                                   const __bf16* __restrict__ Kb,
                                                   const __bf16* __restrict__ Vt,
                                                   const float* __restrict__ maskT,
                                                   __bf16* __restrict__ attnb) {
  __shared__ __align__(32) __bf16 lds[8 * 16 * 64];   // per-wave 16x64 P tile
  const int wid  = threadIdx.x >> 5;
  const int lane = threadIdx.x & 31;
  const int lrow = lane & 15;
  const int hi   = lane >> 4;
  const int t    = blockIdx.x * 8 + wid;   // 4096 wave tiles
  const int bh   = t >> 7;                 // 0..31
  const int q0   = (t & 127) << 4;
  const int b    = bh >> 4;
  const int h    = bh & 15;

  __bf16* P = lds + wid * (16 * 64);

  // Hoist the Q A-fragments (d = 0..31 and 32..63); Q is pre-scaled by 1/sqrt(dk).
  const __bf16* qbase = Qb + ((size_t)bh * SEQ + (q0 + lrow)) * D_K;
  bf16x16 aq0 = make_a_frag(qbase + hi * 16,      qbase + 8 + hi * 16);
  bf16x16 aq1 = make_a_frag(qbase + 32 + hi * 16, qbase + 40 + hi * 16);

  float mrow[8], lsum[8];                   // lsum is a per-lane partial sum
#pragma unroll
  for (int r = 0; r < 8; ++r) { mrow[r] = -1e30f; lsum[r] = 0.f; }
  f32x8 o[4] = {};

  const __bf16* kbase = Kb + (size_t)bh * SEQ * D_K;
  const __bf16* vbase = Vt + (size_t)bh * D_K * SEQ;
  const float*  mbase = maskT + q0 + hi * 8;

  for (int kt = 0; kt < SEQ; kt += 64) {
    // ---- scores: S = Q K^T + maskbias over 64 keys (4 accumulators, 8 WMMAs).
    //      The mask bias vector load IS the C fragment -> no VALU add needed.
    f32x8 sc[4];
#pragma unroll
    for (int j2 = 0; j2 < 4; ++j2) {
      const __bf16* kp = kbase + (size_t)(kt + j2 * 16 + lrow) * D_K + hi * 16;
      bf16x16 b0 = *(const bf16x16*)(kp);
      bf16x16 b1 = *(const bf16x16*)(kp + 32);
      f32x8 mb = *(const f32x8*)(mbase + (size_t)(kt + j2 * 16 + lrow) * SEQ);
      sc[j2] = wmma_bf16(aq0, b0, mb);
      sc[j2] = wmma_bf16(aq1, b1, sc[j2]);
    }

    // ---- online softmax; rows live across the 16-lane halves ----
#pragma unroll
    for (int r = 0; r < 8; ++r) {
      float v0 = sc[0][r];
      float v1 = sc[1][r];
      float v2 = sc[2][r];
      float v3 = sc[3][r];
      float tmax = fmaxf(fmaxf(v0, v1), fmaxf(v2, v3));
#pragma unroll
      for (int off = 1; off < 16; off <<= 1)
        tmax = fmaxf(tmax, __shfl_xor(tmax, off, 32));
      float mnew  = fmaxf(mrow[r], tmax);
      float alpha = __expf(mrow[r] - mnew);
      mrow[r] = mnew;
      float p0 = __expf(v0 - mnew);
      float p1 = __expf(v1 - mnew);
      float p2 = __expf(v2 - mnew);
      float p3 = __expf(v3 - mnew);
      lsum[r] = lsum[r] * alpha + (p0 + p1 + p2 + p3);   // lane-partial; reduced once at end
#pragma unroll
      for (int j = 0; j < 4; ++j) o[j][r] *= alpha;
      __bf16* pr = P + (r + hi * 8) * 64 + lrow;
      pr[0]  = (__bf16)p0;
      pr[16] = (__bf16)p1;
      pr[32] = (__bf16)p2;
      pr[48] = (__bf16)p3;
    }

    // ---- O += P * V : P read back as two A fragments (keys 0..31, 32..63) ----
    bf16x16 ap0 = make_a_frag(P + lrow * 64 + hi * 16,      P + lrow * 64 + 8 + hi * 16);
    bf16x16 ap1 = make_a_frag(P + lrow * 64 + 32 + hi * 16, P + lrow * 64 + 40 + hi * 16);
#pragma unroll
    for (int j = 0; j < 4; ++j) {
      const __bf16* vp = vbase + (size_t)(j * 16 + lrow) * SEQ + kt + hi * 16;
      bf16x16 bv0 = *(const bf16x16*)(vp);
      bf16x16 bv1 = *(const bf16x16*)(vp + 32);
      o[j] = wmma_bf16(ap0, bv0, o[j]);
      o[j] = wmma_bf16(ap1, bv1, o[j]);
    }
  }

  // ---- final row-sum reduction (once), normalize, store [b, s, h, dk] ----
#pragma unroll
  for (int r = 0; r < 8; ++r) {
    float s = lsum[r];
#pragma unroll
    for (int off = 1; off < 16; off <<= 1)
      s += __shfl_xor(s, off, 32);
    float inv = 1.f / s;
    const int q = q0 + r + hi * 8;
#pragma unroll
    for (int j = 0; j < 4; ++j) {
      const int dk = j * 16 + lrow;
      attnb[((size_t)(b * SEQ + q) * N_HEADS + h) * D_K + dk] = (__bf16)(o[j][r] * inv);
    }
  }
}

// ----------------------------------------------------------------- launcher
extern "C" void kernel_launch(void* const* d_in, const int* in_sizes, int n_in,
                              void* d_out, int out_size, void* d_ws, size_t ws_size,
                              hipStream_t stream) {
  (void)in_sizes; (void)n_in; (void)out_size; (void)ws_size;
  const float* X    = (const float*)d_in[0];
  const int*   mask = (const int*)d_in[1];
  const float* Wq   = (const float*)d_in[2];
  const float* bq   = (const float*)d_in[3];
  const float* Wk   = (const float*)d_in[4];
  const float* bk   = (const float*)d_in[5];
  const float* Wv   = (const float*)d_in[6];
  const float* bv   = (const float*)d_in[7];
  const float* Wo   = (const float*)d_in[8];
  const float* bo   = (const float*)d_in[9];

  char* ws = (char*)d_ws;
  size_t off = 0;
  auto alloc = [&](size_t bytes) -> void* {
    void* p = ws + off;
    off += (bytes + 255) & ~(size_t)255;
    return p;
  };
  const size_t NX = (size_t)M_TOT * D_MODEL;     // 4M elements
  const size_t NW = (size_t)D_MODEL * D_MODEL;   // 1M elements
  const size_t NM = (size_t)SEQ * SEQ;           // 4M elements
  __bf16* Xb    = (__bf16*)alloc(NX * 2);
  __bf16* Wqb   = (__bf16*)alloc(NW * 2);
  __bf16* Wkb   = (__bf16*)alloc(NW * 2);
  __bf16* Wvb   = (__bf16*)alloc(NW * 2);
  __bf16* Wob   = (__bf16*)alloc(NW * 2);
  __bf16* Qb    = (__bf16*)alloc(NX * 2);
  __bf16* Kb    = (__bf16*)alloc(NX * 2);
  __bf16* Vt    = (__bf16*)alloc(NX * 2);
  __bf16* Attnb = (__bf16*)alloc(NX * 2);
  float*  maskT = (float*)alloc(NM * 4);

  // staging
  cvt_bf16_kernel<<<(int)(NX / 256), 256, 0, stream>>>(X,  Xb,  (int)NX);
  cvt_bf16_kernel<<<(int)(NW / 256), 256, 0, stream>>>(Wq, Wqb, (int)NW);
  cvt_bf16_kernel<<<(int)(NW / 256), 256, 0, stream>>>(Wk, Wkb, (int)NW);
  cvt_bf16_kernel<<<(int)(NW / 256), 256, 0, stream>>>(Wv, Wvb, (int)NW);
  cvt_bf16_kernel<<<(int)(NW / 256), 256, 0, stream>>>(Wo, Wob, (int)NW);
  maskbias_kernel<<<(int)(NM / 256), 256, 0, stream>>>(mask, maskT);

  // QKV projections (WMMA GEMM); Q pre-scaled; V stored transposed
  proj_kernel<0><<<256, 256, 0, stream>>>(Xb, Wqb, bq, Qb);
  proj_kernel<1><<<256, 256, 0, stream>>>(Xb, Wkb, bk, Kb);
  proj_kernel<2><<<256, 256, 0, stream>>>(Xb, Wvb, bv, Vt);

  // flash attention
  attn_kernel<<<512, 256, 0, stream>>>(Qb, Kb, Vt, maskT, Attnb);

  // output projection, fp32 result
  proj_kernel<3><<<256, 256, 0, stream>>>(Attnb, Wob, bo, d_out);
}
